// Text2Motion_Transformer_Word_CrossAtt_21414706938510
// MI455X (gfx1250) — matrix operand order repaired and checked by reading, compiled, not run on hardware
//
#include <hip/hip_runtime.h>

// ---------------------------------------------------------------------------
// CDNA5 (gfx1250, wave32) mixed local/global attention block.
// Matrix work: v_wmma_f32_16x16x32_bf16.
// Shared tiles staged via gfx1250 async DMA: global_load_async_to_lds_b128
// (ASYNCcnt / s_wait_asynccnt), double-buffered in LDS. Fragment loads are
// batched ahead of each WMMA chain so dscnt waits can be partial.
// ---------------------------------------------------------------------------

typedef __attribute__((ext_vector_type(16))) __bf16        v16bf;
typedef __attribute__((ext_vector_type(8)))  float         v8f;
typedef __attribute__((ext_vector_type(4)))  unsigned int  u32x4;

union V16 { v16bf bf; u32x4 q[2]; };

#define T_SEQ 1024     // 2*BLOCK + 2
#define BLK   511
#define HD    64
#define EMB   512
#define NVH   10       // 8 main heads + 2 aml heads

__device__ __forceinline__ unsigned short f2bf(float f) {
  union { float f; unsigned u; } v; v.f = f;
  unsigned u = v.u;
  u += 0x7FFFu + ((u >> 16) & 1u);          // round-to-nearest-even
  return (unsigned short)(u >> 16);
}

// LDS byte offset of a __shared__ object: generic-pointer low 32 bits
// (ISA 10.2 aperture mapping).
__device__ __forceinline__ unsigned lds_off(const void* p) {
  return (unsigned)(uintptr_t)p;
}

// gfx1250 async DMA: copy 16 bytes global -> LDS, tracked by ASYNCcnt.
__device__ __forceinline__ void async_b128(unsigned loff, const void* g) {
  asm volatile("global_load_async_to_lds_b128 %0, %1, off"
               :: "v"(loff), "v"(g) : "memory");
}
__device__ __forceinline__ void wait_async0() {
  asm volatile("s_wait_asynccnt 0x0" ::: "memory");
}

__global__ void cast_f32_bf16(const float* __restrict__ s,
                              unsigned short* __restrict__ d, int n) {
  int i = blockIdx.x * blockDim.x + threadIdx.x;
  if (i < n) d[i] = f2bf(s[i]);
}

// ---------------------------------------------------------------------------
// WMMA GEMM: out = X (N x C, bf16) @ W^T (W is O x C, bf16) + bias.
// 4 waves / block, block tile 64 rows x 64 cols, K step 32.
// The 64x32 W tile is shared by all 4 waves -> staged once per block into LDS
// by async DMA, double-buffered (prefetch k0+32 while computing k0).
// mode 0: bf16 head-major  out[((b*nh+h)*T + t)*HD + d]
// mode 1: bf16 head-major transposed (for V): out[((b*nh+h)*HD + d)*T + t]
// mode 2: f32 row-major out[r*O + c]  (final projection -> d_out)
// ---------------------------------------------------------------------------
__global__ __launch_bounds__(128)
void gemm_bf16_wmma(const unsigned short* __restrict__ X,
                    const unsigned short* __restrict__ W,
                    const float* __restrict__ bias,
                    unsigned short* __restrict__ outb,
                    float* __restrict__ outf,
                    int C, int O, int nh, int mode) {
  __shared__ unsigned short sW[2][64 * 32];   // 2 x 4KB, [col][k]

  const int tid  = threadIdx.x;
  const int lane = tid & 31;
  const int wv   = tid >> 5;
  const int l16  = lane & 15;
  const int half = lane >> 4;
  const int row0 = blockIdx.y * 64 + wv * 16;
  const int col0 = blockIdx.x * 64;

  // cooperative stage of one 64x32 W tile: 256 x 16B chunks, 2 per thread
  auto stage = [&](int k0, int buf) {
#pragma unroll
    for (int r = 0; r < 2; ++r) {
      int idx = tid + r * 128;
      int col = idx >> 2;
      int kc  = (idx & 3) * 8;
      async_b128(lds_off(&sW[buf][col * 32 + kc]),
                 W + (size_t)(col0 + col) * C + k0 + kc);
    }
  };

  v8f acc[4];
#pragma unroll
  for (int n = 0; n < 4; ++n) acc[n] = (v8f){0,0,0,0,0,0,0,0};

  stage(0, 0);
  for (int k0 = 0; k0 < C; k0 += 32) {
    const int buf = (k0 >> 5) & 1;
    wait_async0();           // my async transfers into sW[buf] are done
    __syncthreads();         // everyone's are
    if (k0 + 32 < C) stage(k0 + 32, buf ^ 1);

    // A fragment (16x32) from global: lane = row, K contiguous 16*half..+15
    V16 a;
    const unsigned short* ap = X + (size_t)(row0 + l16) * C + k0;
    a.q[0] = *(const u32x4*)(ap + 16 * half);
    a.q[1] = *(const u32x4*)(ap + 16 * half + 8);

    // batch all four B fragments (32x16 each) from LDS first
    V16 bfrag[4];
#pragma unroll
    for (int n = 0; n < 4; ++n) {
      const unsigned short* bp = &sW[buf][(n * 16 + l16) * 32 + 16 * half];
      bfrag[n].q[0] = *(const u32x4*)(bp);
      bfrag[n].q[1] = *(const u32x4*)(bp + 8);
    }
    // then the WMMA chain (A reused 4x)
#pragma unroll
    for (int n = 0; n < 4; ++n)
      acc[n] = __builtin_amdgcn_wmma_f32_16x16x32_bf16(
          false, a.bf, false, bfrag[n].bf, (short)0, acc[n], false, false);

    __syncthreads();         // all waves done reading sW[buf] before it is reused
  }

#pragma unroll
  for (int n = 0; n < 4; ++n) {
#pragma unroll
    for (int j = 0; j < 8; ++j) {
      int r = row0 + j + 8 * half;          // D layout: M = j + 8*half
      int c = col0 + n * 16 + l16;          //           N = lane & 15
      float v = acc[n][j] + bias[c];
      if (mode == 2) {
        outf[(size_t)r * O + c] = v;
      } else {
        int h = c >> 6, d = c & 63;
        int b = r / T_SEQ, t = r % T_SEQ;
        size_t idx = (mode == 0)
            ? ((size_t)(b * nh + h) * T_SEQ + t) * HD + d
            : ((size_t)(b * nh + h) * HD + d) * (size_t)T_SEQ + t;
        outb[idx] = f2bf(v);
      }
    }
  }
}

// ---------------------------------------------------------------------------
// Mask math replicating _build_masks() elementwise.
// ---------------------------------------------------------------------------
__device__ __forceinline__ int temporal_i(int i) {
  return (i < 2) ? 0 : ((i < 2 + BLK) ? (2 * (i - 2) + 1) : (2 * (i - 2 - BLK) + 2));
}
__device__ __forceinline__ bool visible(int q, int k, bool local) {
  bool g = temporal_i(k) <= temporal_i(q);
  if (!local || !g) return g;
  if (q >= 2 && k >= 2) {
    int tq = 2 * ((q - 2) % BLK) - 6;       // 2*r+1 - 2*RECEP + 1, RECEP=4
    int tk = 2 * ((k - 2) % BLK) + 1;
    if (tk < tq) return false;              // key outside sliding window
  }
  return true;
}

// ---------------------------------------------------------------------------
// Flash attention over 10 virtual heads. 4 waves share one (b, vh) and split
// 64 q rows; K and V^T tiles are shared -> async-staged into LDS, double
// buffered. Per 32-key step and wave: 4 WMMAs for S (K=64 chained), online
// softmax with 16-lane shuffles, P staged via LDS to A-layout, 4 WMMAs P@V.
// ---------------------------------------------------------------------------
__global__ __launch_bounds__(128)
void attn_flash_wmma(const unsigned short* __restrict__ q_ws,
                     const unsigned short* __restrict__ k_ws,
                     const unsigned short* __restrict__ vt_ws,
                     const unsigned short* __restrict__ qml_ws,
                     const unsigned short* __restrict__ kml_ws,
                     const float* __restrict__ bias_clip,
                     const float* __restrict__ bias_clip_ml,
                     float* __restrict__ y_ws, int B) {
  __shared__ unsigned short sK[2][32 * 64];     // [key][d]   2 x 4KB
  __shared__ unsigned short sV[2][64 * 32];     // [d][key]   2 x 4KB
  __shared__ unsigned short plds[4][16 * 32];   // per-wave P staging

  const int tid  = threadIdx.x;
  const int lane = tid & 31;
  const int wv   = tid >> 5;
  const int l16  = lane & 15;
  const int half = lane >> 4;
  const int q0   = blockIdx.x * 64 + wv * 16;
  const int vh   = blockIdx.y % NVH;
  const int b    = blockIdx.y / NVH;

  const unsigned short *qp, *kp, *vp;
  float biash; bool local;
  if (vh < 8) {
    size_t o = (size_t)(b * 8 + vh) * T_SEQ * HD;
    qp = q_ws + o; kp = k_ws + o; vp = vt_ws + o;
    biash = bias_clip[vh]; local = (vh < 2);
  } else {
    int m = vh - 8;
    size_t oml = (size_t)(b * 2 + m) * T_SEQ * HD;
    qp = qml_ws + oml; kp = kml_ws + oml;
    vp = vt_ws + (size_t)(b * 8 + 2 + m) * T_SEQ * HD;  // mix heads use V of head 2+m
    biash = bias_clip_ml[m]; local = true;
  }

  // cooperative stage of K (32x64) and V^T (64x32) tiles: 2+2 chunks / thread
  auto stage = [&](int kt, int buf) {
#pragma unroll
    for (int r = 0; r < 2; ++r) {
      int idx = tid + r * 128;
      int krow = idx >> 3, kd = (idx & 7) * 8;          // K: row, d-chunk
      async_b128(lds_off(&sK[buf][krow * 64 + kd]),
                 kp + (size_t)(kt + krow) * HD + kd);
      int drow = idx >> 2, kk = (idx & 3) * 8;          // V^T: d-row, key-chunk
      async_b128(lds_off(&sV[buf][drow * 32 + kk]),
                 vp + (size_t)drow * T_SEQ + kt + kk);
    }
  };

  // Q A-fragments for d=0..31 and d=32..63 (loaded once, from global)
  V16 aq0, aq1;
  {
    const unsigned short* r = qp + (size_t)(q0 + l16) * HD;
    aq0.q[0] = *(const u32x4*)(r + 16 * half);
    aq0.q[1] = *(const u32x4*)(r + 16 * half + 8);
    aq1.q[0] = *(const u32x4*)(r + 32 + 16 * half);
    aq1.q[1] = *(const u32x4*)(r + 32 + 16 * half + 8);
  }

  float m_[8], l_[8];
  v8f oacc[4];
#pragma unroll
  for (int j = 0; j < 8; ++j) { m_[j] = -3.0e38f; l_[j] = 0.0f; }
#pragma unroll
  for (int n = 0; n < 4; ++n) oacc[n] = (v8f){0,0,0,0,0,0,0,0};

  const float scale = 0.125f;               // 1/sqrt(HD)

  stage(0, 0);
  for (int kt = 0; kt < T_SEQ; kt += 32) {
    const int buf = (kt >> 5) & 1;
    wait_async0();
    __syncthreads();
    if (kt + 32 < T_SEQ) stage(kt + 32, buf ^ 1);

    // --- batch all K fragments for both 16-key tiles, then 4 WMMAs ---
    V16 bk[2][2];
#pragma unroll
    for (int t = 0; t < 2; ++t) {
      const unsigned short* kr = &sK[buf][(t * 16 + l16) * 64];
      bk[t][0].q[0] = *(const u32x4*)(kr + 16 * half);
      bk[t][0].q[1] = *(const u32x4*)(kr + 16 * half + 8);
      bk[t][1].q[0] = *(const u32x4*)(kr + 32 + 16 * half);
      bk[t][1].q[1] = *(const u32x4*)(kr + 32 + 16 * half + 8);
    }
    v8f s[2];
#pragma unroll
    for (int t = 0; t < 2; ++t) {
      v8f z = (v8f){0,0,0,0,0,0,0,0};
      s[t] = __builtin_amdgcn_wmma_f32_16x16x32_bf16(
          false, aq0.bf, false, bk[t][0].bf, (short)0, z, false, false);
      s[t] = __builtin_amdgcn_wmma_f32_16x16x32_bf16(
          false, aq1.bf, false, bk[t][1].bf, (short)0, s[t], false, false);
    }

    // --- scale + CLIP-key bias + mask + online softmax ---
    float pv[2][8];
#pragma unroll
    for (int j = 0; j < 8; ++j) {
      int qrow = q0 + j + 8 * half;
      float sv[2], rm = -3.0e38f;
#pragma unroll
      for (int t = 0; t < 2; ++t) {
        int kc = kt + t * 16 + l16;
        float v = s[t][j] * scale;
        if (kc == 1) v += biash;
        v = visible(qrow, kc, local) ? v : -1.0e9f;
        sv[t] = v;
        rm = fmaxf(rm, v);
      }
      for (int off = 1; off < 16; off <<= 1)        // row max over 16-lane group
        rm = fmaxf(rm, __shfl_xor(rm, off, 32));
      float nm    = fmaxf(m_[j], rm);
      float alpha = __expf(m_[j] - nm);
      float rs = 0.0f;
#pragma unroll
      for (int t = 0; t < 2; ++t) { float p = __expf(sv[t] - nm); pv[t][j] = p; rs += p; }
      for (int off = 1; off < 16; off <<= 1)        // row sum
        rs += __shfl_xor(rs, off, 32);
      l_[j] = l_[j] * alpha + rs;
      m_[j] = nm;
#pragma unroll
      for (int n = 0; n < 4; ++n) oacc[n][j] *= alpha;
    }

    // --- stage P (16x32) to LDS, reload in A-fragment layout ---
#pragma unroll
    for (int j = 0; j < 8; ++j) {
      int row = j + 8 * half;
      plds[wv][row * 32 + l16]      = f2bf(pv[0][j]);
      plds[wv][row * 32 + 16 + l16] = f2bf(pv[1][j]);
    }
    __syncthreads();
    V16 pa;
    {
      const unsigned short* pr = &plds[wv][l16 * 32];
      pa.q[0] = *(const u32x4*)(pr + 16 * half);
      pa.q[1] = *(const u32x4*)(pr + 16 * half + 8);
    }

    // --- batch all V fragments, then 4 WMMAs: O += P @ V ---
    V16 vb[4];
#pragma unroll
    for (int n = 0; n < 4; ++n) {
      const unsigned short* vr = &sV[buf][(n * 16 + l16) * 32 + 16 * half];
      vb[n].q[0] = *(const u32x4*)(vr);
      vb[n].q[1] = *(const u32x4*)(vr + 8);
    }
#pragma unroll
    for (int n = 0; n < 4; ++n)
      oacc[n] = __builtin_amdgcn_wmma_f32_16x16x32_bf16(
          false, pa.bf, false, vb[n].bf, (short)0, oacc[n], false, false);

    __syncthreads();   // done reading sK/sV/plds[buf] before next overwrite
  }

  // --- epilogue: normalize, store f32 per virtual head ---
  float* yo = y_ws + ((size_t)vh * B + b) * T_SEQ * HD;
#pragma unroll
  for (int j = 0; j < 8; ++j) {
    int row = q0 + j + 8 * half;
    float inv = 1.0f / l_[j];
#pragma unroll
    for (int n = 0; n < 4; ++n)
      yo[(size_t)row * HD + n * 16 + l16] = oacc[n][j] * inv;
  }
}

// ---------------------------------------------------------------------------
// Blend: heads 2,3 = mix_w[h][0]*global + mix_w[h][1]*aml; cast to bf16,
// repack to (B*T, 512) row-major for the output projection.
// ---------------------------------------------------------------------------
__global__ void blend_heads(const float* __restrict__ y_ws,
                            const float* __restrict__ mixw,
                            unsigned short* __restrict__ ybf, int B) {
  int idx = blockIdx.x * blockDim.x + threadIdx.x;
  int total = B * T_SEQ * EMB;
  if (idx >= total) return;
  int d = idx & 63;
  int h = (idx >> 6) & 7;
  int t = (idx >> 9) % T_SEQ;
  int b = idx / (T_SEQ * EMB);
  size_t base = (((size_t)h * B + b) * T_SEQ + t) * HD + d;
  float val;
  if (h >= 2 && h < 4) {
    int m = h - 2;
    float yg = y_ws[base];
    float yl = y_ws[(((size_t)(8 + m) * B + b) * T_SEQ + t) * HD + d];
    val = mixw[m * 2 + 0] * yg + mixw[m * 2 + 1] * yl;
  } else {
    val = y_ws[base];
  }
  ybf[((size_t)(b * T_SEQ + t)) * EMB + h * 64 + d] = f2bf(val);
}

// ---------------------------------------------------------------------------
// Host side
// ---------------------------------------------------------------------------
extern "C" void kernel_launch(void* const* d_in, const int* in_sizes, int n_in,
                              void* d_out, int out_size, void* d_ws, size_t ws_size,
                              hipStream_t stream) {
  const float* x        = (const float*)d_in[0];
  const float* Wq       = (const float*)d_in[2];
  const float* bq       = (const float*)d_in[3];
  const float* Wk       = (const float*)d_in[4];
  const float* bk       = (const float*)d_in[5];
  const float* Wv       = (const float*)d_in[6];
  const float* bv       = (const float*)d_in[7];
  const float* Wqml     = (const float*)d_in[8];
  const float* bqml     = (const float*)d_in[9];
  const float* Wkml     = (const float*)d_in[10];
  const float* bkml     = (const float*)d_in[11];
  const float* mix_w    = (const float*)d_in[12];
  const float* Wp       = (const float*)d_in[13];
  const float* bp       = (const float*)d_in[14];
  const float* biasclip = (const float*)d_in[15];
  const float* biasml   = (const float*)d_in[16];

  const int B = in_sizes[0] / (T_SEQ * EMB);   // 4
  const int N = B * T_SEQ;                     // 4096

  char* ws = (char*)d_ws;
  size_t off = 0;
  auto take = [&](size_t bytes) { char* p = ws + off; off += (bytes + 255) & ~(size_t)255; return p; };
  unsigned short* x_bf   = (unsigned short*)take((size_t)N * EMB * 2);
  unsigned short* wq_bf  = (unsigned short*)take((size_t)EMB * EMB * 2);
  unsigned short* wk_bf  = (unsigned short*)take((size_t)EMB * EMB * 2);
  unsigned short* wv_bf  = (unsigned short*)take((size_t)EMB * EMB * 2);
  unsigned short* wqm_bf = (unsigned short*)take((size_t)128 * EMB * 2);
  unsigned short* wkm_bf = (unsigned short*)take((size_t)128 * EMB * 2);
  unsigned short* wp_bf  = (unsigned short*)take((size_t)EMB * EMB * 2);
  unsigned short* q_ws   = (unsigned short*)take((size_t)B * 8 * T_SEQ * HD * 2);
  unsigned short* k_ws   = (unsigned short*)take((size_t)B * 8 * T_SEQ * HD * 2);
  unsigned short* vt_ws  = (unsigned short*)take((size_t)B * 8 * T_SEQ * HD * 2);
  unsigned short* qml_ws = (unsigned short*)take((size_t)B * 2 * T_SEQ * HD * 2);
  unsigned short* kml_ws = (unsigned short*)take((size_t)B * 2 * T_SEQ * HD * 2);
  float*          y_ws   = (float*)take((size_t)NVH * B * T_SEQ * HD * 4);
  unsigned short* ybf_ws = (unsigned short*)take((size_t)N * EMB * 2);

  // 1) casts to bf16
  auto cast = [&](const float* s, unsigned short* d, int n) {
    cast_f32_bf16<<<(n + 255) / 256, 256, 0, stream>>>(s, d, n);
  };
  cast(x,    x_bf,   N * EMB);
  cast(Wq,   wq_bf,  EMB * EMB);
  cast(Wk,   wk_bf,  EMB * EMB);
  cast(Wv,   wv_bf,  EMB * EMB);
  cast(Wqml, wqm_bf, 128 * EMB);
  cast(Wkml, wkm_bf, 128 * EMB);
  cast(Wp,   wp_bf,  EMB * EMB);

  // 2) projections (WMMA GEMMs, async-LDS staged weight tiles)
  dim3 blk(128);
  dim3 g512(EMB / 64, N / 64), g128(128 / 64, N / 64);
  gemm_bf16_wmma<<<g512, blk, 0, stream>>>(x_bf, wq_bf,  bq,   q_ws,   nullptr, EMB, EMB, 8, 0);
  gemm_bf16_wmma<<<g512, blk, 0, stream>>>(x_bf, wk_bf,  bk,   k_ws,   nullptr, EMB, EMB, 8, 0);
  gemm_bf16_wmma<<<g512, blk, 0, stream>>>(x_bf, wv_bf,  bv,   vt_ws,  nullptr, EMB, EMB, 8, 1);
  gemm_bf16_wmma<<<g128, blk, 0, stream>>>(x_bf, wqm_bf, bqml, qml_ws, nullptr, EMB, 128, 2, 0);
  gemm_bf16_wmma<<<g128, blk, 0, stream>>>(x_bf, wkm_bf, bkml, kml_ws, nullptr, EMB, 128, 2, 0);

  // 3) flash attention over 10 virtual heads (async-LDS staged K/V tiles)
  attn_flash_wmma<<<dim3(T_SEQ / 64, NVH * B), blk, 0, stream>>>(
      q_ws, k_ws, vt_ws, qml_ws, kml_ws, biasclip, biasml, y_ws, B);

  // 4) blend mix heads, repack to (B*T, 512) bf16
  int tot = B * T_SEQ * EMB;
  blend_heads<<<(tot + 255) / 256, 256, 0, stream>>>(y_ws, mix_w, ybf_ws, B);

  // 5) output projection -> f32 d_out
  gemm_bf16_wmma<<<g512, blk, 0, stream>>>(ybf_ws, wp_bf, bp, nullptr, (float*)d_out,
                                           EMB, EMB, 1, 2);
}
